// DSTCarbonFormer_27109833573027
// MI455X (gfx1250) — matrix operand order repaired and checked by reading, not compiled, not run on hardware
//
#include <hip/hip_runtime.h>
#include <hip/hip_bf16.h>

// ---------------- problem constants ----------------
#define DIMC 64
#define BB   2
#define TT   4
#define HH   64
#define WW   64
#define SS   (TT*HH*WW)       // 16384 voxels per batch
#define NTOT (BB*SS)          // 32768 voxels total
#define KCONV (27*DIMC)       // 1728 = implicit-GEMM K for 3x3x3, Cin=64
#define NE_  3
#define DI_  128
#define DS_  16
#define DR_  4

typedef __attribute__((ext_vector_type(16))) __bf16          bf16x16;
typedef __attribute__((ext_vector_type(8)))  float           f32x8;
typedef __attribute__((ext_vector_type(8)))  unsigned short  us8;
typedef __attribute__((ext_vector_type(4)))  unsigned int    u32x4;
typedef __attribute__((ext_vector_type(8)))  int             i32x8;
typedef __attribute__((ext_vector_type(4)))  int             i32x4;

union ABPack { bf16x16 v; us8 h[2]; };

#if defined(__has_builtin)
# if __has_builtin(__builtin_amdgcn_tensor_load_to_lds)
#  define HAVE_TDM 1
# endif
#endif
#ifndef HAVE_TDM
# define HAVE_TDM 0
#endif

__device__ __forceinline__ unsigned short f2bf(float f) {
  unsigned u = __builtin_bit_cast(unsigned, f);
  unsigned r = u + 0x7fffu + ((u >> 16) & 1u);
  return (unsigned short)(r >> 16);
}
__device__ __forceinline__ float siluf(float x) { return x / (1.f + __expf(-x)); }
__device__ __forceinline__ float softplusf(float x) {
  return (x > 20.f) ? x : log1pf(__expf(x));
}

// =====================================================================
// Weight packing: f32 OIDHW -> bf16 [co][tap*Cin + ci]
// =====================================================================
__global__ void k_prep_convw(const float* __restrict__ w, unsigned short* __restrict__ wb,
                             int Cout, int Cin) {
  long total = (long)Cout * 27 * Cin;
  for (long i = (long)blockIdx.x * blockDim.x + threadIdx.x; i < total;
       i += (long)gridDim.x * blockDim.x) {
    int ci  = (int)(i % Cin);
    long r  = i / Cin;
    int tap = (int)(r % 27);
    int co  = (int)(r / 27);
    float v = w[((long)co * Cin + ci) * 27 + tap];
    wb[(long)co * (27 * Cin) + (long)tap * Cin + ci] = f2bf(v);
  }
}

// f32 [Cout][Cin] -> bf16 [CoutPad][Cin], zero pad rows
__global__ void k_prep_densew(const float* __restrict__ w, unsigned short* __restrict__ wb,
                              int Cout, int Cin, int CoutPad) {
  long total = (long)CoutPad * Cin;
  for (long i = (long)blockIdx.x * blockDim.x + threadIdx.x; i < total;
       i += (long)gridDim.x * blockDim.x) {
    int ci = (int)(i % Cin);
    int co = (int)(i / Cin);
    wb[i] = (co < Cout) ? f2bf(w[(long)co * Cin + ci]) : (unsigned short)0;
  }
}

// =====================================================================
// WMMA implicit-GEMM 3x3x3 conv, Cin=64.
//   xbt : bf16 activations, voxel-major [NTOT][64]
//   wb  : bf16 weights [CoutTotal][1728]  (k = tap*64 + ci)
//   out : f32 [b][CoutStore][S]
// block = 128 threads = 4 waves; each wave -> one 16(co) x 16(vox) tile.
// Weight slab for this workgroup's 64 output channels staged in LDS via TDM.
// =====================================================================
__global__ __launch_bounds__(128)
void k_conv3x3_wmma(const unsigned short* __restrict__ xbt,
                    const unsigned short* __restrict__ wb,
                    const float* __restrict__ bias,
                    float* __restrict__ out,
                    int CoutStore, int dil) {
  extern __shared__ unsigned short wlds[];   // 64 * 1728 bf16 = 216 KB
  const int tid  = threadIdx.x;
  const int lane = tid & 31;
  const int wid  = tid >> 5;
  const int coG  = blockIdx.y * 64;

  const unsigned short* wsrc = wb + (size_t)coG * KCONV;

#if HAVE_TDM
  if (wid == 0) {
    unsigned lds_off = (unsigned)(unsigned long long)(void*)wlds;
    unsigned long long ga = (unsigned long long)(const void*)wsrc;
    unsigned nelem = (64u * KCONV * 2u) / 8u;   // 8-byte elements: 27648
    u32x4 g0;
    g0[0] = 1u;                                  // count = 1 (valid user descriptor)
    g0[1] = lds_off;                             // lds_addr
    g0[2] = (unsigned)(ga & 0xffffffffu);        // global_addr[31:0]
    g0[3] = (unsigned)((ga >> 32) & 0x1ffffffu) | (2u << 30);  // addr[56:32] | type=2
    i32x8 g1 = {0,0,0,0,0,0,0,0};
    g1[0] = (int)(3u << 16);                     // workgroup_mask=0, data_size=8B
    g1[1] = (int)((nelem & 0xffffu) << 16);      // tensor_dim0[15:0]  @ bits 63:48
    g1[2] = (int)((nelem >> 16) | (1u << 16));   // tensor_dim0[31:16] | tensor_dim1=1
    g1[3] = (int)(nelem << 16);                  // tile_dim0 = nelem  @ bits 127:112
    g1[4] = 0;                                   // tile_dim1=0 (unused), tile_dim2=0
    g1[5] = (int)nelem;                          // tensor_dim0_stride low bits
    g1[6] = 0;
    g1[7] = 0;
    i32x4 z4 = {0,0,0,0};
    i32x8 zg3 = {0,0,0,0,0,0,0,0};
    __builtin_amdgcn_tensor_load_to_lds(g0, g1, z4, z4, zg3, 0);
    __builtin_amdgcn_s_wait_tensorcnt(0);
  }
#else
  for (int i = tid * 8; i < 64 * KCONV; i += 128 * 8)
    *(us8*)(wlds + i) = *(const us8*)(wsrc + i);
#endif
  __syncthreads();

  // tile decode: 16 consecutive w positions
  const int vt = blockIdx.x;                 // 0..2047
  const int w0 = (vt & 3) << 4;
  const int h  = (vt >> 2) & 63;
  const int t  = (vt >> 8) & 3;
  const int b  = vt >> 10;

  const int  mrow  = lane & 15;
  const bool hihalf = lane >= 16;
  const int  abase = hihalf ? 16 : 0;        // A-operand K-half base
  const int  bkoff = hihalf ? 16 : 0;        // B-operand K offset
  const int  colw  = w0 + mrow;              // this lane's B-column voxel w

  __builtin_prefetch(xbt + (((long)(b * TT + t) * HH + h) * WW + w0) * 64, 0, 1);

  f32x8 acc = {0.f,0.f,0.f,0.f,0.f,0.f,0.f,0.f};
  const us8 z8 = {0,0,0,0,0,0,0,0};

  for (int kd = 0; kd < 3; ++kd) {
    int tn = t + (kd - 1) * dil;
    if (tn < 0 || tn >= TT) continue;
    for (int kh = 0; kh < 3; ++kh) {
      int hn = h + (kh - 1) * dil;
      if (hn < 0 || hn >= HH) continue;
      long nb = ((long)(b * TT + tn) * HH + hn) * WW;
      for (int kw = 0; kw < 3; ++kw) {
        int wn = colw + (kw - 1) * dil;
        bool valid = (wn >= 0) && (wn < WW);
        int  wc    = valid ? wn : 0;
        const unsigned short* bp = xbt + (nb + wc) * 64;
        int tap   = (kd * 3 + kh) * 3 + kw;
        int kbase = tap * 64;
#pragma unroll
        for (int ks = 0; ks < 2; ++ks) {
          ABPack A, Bv;
          const unsigned short* ap =
              wlds + ((long)(wid * 16 + mrow) * KCONV + kbase + ks * 32 + abase);
          A.h[0] = *(const us8*)ap;
          A.h[1] = *(const us8*)(ap + 8);
          Bv.v = *(const bf16x16*)(bp + ks * 32 + bkoff);
          if (!valid) { Bv.h[0] = z8; Bv.h[1] = z8; }
          acc = __builtin_amdgcn_wmma_f32_16x16x32_bf16(
              false, A.v, false, Bv.v, (short)0, acc, false, false);
        }
      }
    }
  }

  const int co0 = coG + wid * 16;
  const long sidx = (long)t * HH * WW + (long)h * WW + (w0 + mrow);
#pragma unroll
  for (int r = 0; r < 8; ++r) {
    int co = co0 + r + (hihalf ? 8 : 0);
    float v = acc[r] + (bias ? bias[co] : 0.f);
    out[((long)b * CoutStore + co) * SS + sidx] = v;
  }
}

// =====================================================================
// WMMA dense GEMM (1x1x1 conv / projections)
//   xbt : bf16 [NTOT][Cstride]   wb : bf16 [CoutPad][K]
//   out : f32 [b][CoutStore][S]
// =====================================================================
__global__ __launch_bounds__(128)
void k_gemm_wmma(const unsigned short* __restrict__ xbt, int Cstride,
                 const unsigned short* __restrict__ wb, int K,
                 const float* __restrict__ bias,
                 float* __restrict__ out, int CoutStore, int CoutReal) {
  const int tid = threadIdx.x, lane = tid & 31, wid = tid >> 5;
  const int coG = blockIdx.y * 64 + wid * 16;
  const long n0 = (long)blockIdx.x * 16;
  const int  mrow = lane & 15;
  const bool hihalf = lane >= 16;
  const int  abase = hihalf ? 16 : 0;
  const int  bkoff = hihalf ? 16 : 0;
  const long n = n0 + mrow;

  f32x8 acc = {0.f,0.f,0.f,0.f,0.f,0.f,0.f,0.f};
  for (int k0 = 0; k0 < K; k0 += 32) {
    ABPack A, Bv;
    const unsigned short* ap = wb + ((long)(coG + mrow) * K + k0 + abase);
    A.h[0] = *(const us8*)ap;
    A.h[1] = *(const us8*)(ap + 8);
    Bv.v = *(const bf16x16*)(xbt + n * Cstride + k0 + bkoff);
    acc = __builtin_amdgcn_wmma_f32_16x16x32_bf16(
        false, A.v, false, Bv.v, (short)0, acc, false, false);
  }
  const int  b  = (int)(n0 / SS);
  const long s  = (n0 - (long)b * SS) + mrow;
#pragma unroll
  for (int r = 0; r < 8; ++r) {
    int co = coG + r + (hihalf ? 8 : 0);
    if (co < CoutReal)
      out[((long)b * CoutStore + co) * SS + s] = acc[r] + (bias ? bias[co] : 0.f);
  }
}

// =====================================================================
// Naive 3x3x3 conv for tiny channel counts (heads / 1-channel tail)
// x : f32 [b][Cin][S]   w : f32 [co][ci][27]
// =====================================================================
__global__ void k_conv3_naive(const float* __restrict__ x, const float* __restrict__ w,
                              const float* __restrict__ bias, float* __restrict__ y,
                              int Cin, int Cout, int dil) {
  long total = (long)BB * Cout * SS;
  for (long i = (long)blockIdx.x * blockDim.x + threadIdx.x; i < total;
       i += (long)gridDim.x * blockDim.x) {
    long s = i % SS;
    long r = i / SS;
    int co = (int)(r % Cout);
    int b  = (int)(r / Cout);
    int ww_ = (int)(s % WW);
    int hh_ = (int)((s / WW) % HH);
    int tt_ = (int)(s / (HH * WW));
    float acc = bias[co];
    for (int kd = 0; kd < 3; ++kd) {
      int tn = tt_ + (kd - 1) * dil; if (tn < 0 || tn >= TT) continue;
      for (int kh = 0; kh < 3; ++kh) {
        int hn = hh_ + (kh - 1) * dil; if (hn < 0 || hn >= HH) continue;
        for (int kw = 0; kw < 3; ++kw) {
          int wn = ww_ + (kw - 1) * dil; if (wn < 0 || wn >= WW) continue;
          long sp = (long)tn * HH * WW + (long)hn * WW + wn;
          int tap = (kd * 3 + kh) * 3 + kw;
          for (int ci = 0; ci < Cin; ++ci)
            acc += x[((long)b * Cin + ci) * SS + sp] * w[((long)co * Cin + ci) * 27 + tap];
        }
      }
    }
    y[i] = acc;
  }
}

// =====================================================================
// Generic elementwise: v = act(a[i]) (+ add[i]); optional f32 + bf16(xbt) out
// act: 0 none, 1 prelu(*alphap), 2 leaky 0.1, 3 relu
// =====================================================================
__global__ void k_ew(const float* __restrict__ a, const float* __restrict__ add,
                     float* __restrict__ outf, unsigned short* __restrict__ xbt,
                     int C, int cstride, int coff, int act, const float* __restrict__ alphap) {
  long total = (long)BB * C * SS;
  for (long i = (long)blockIdx.x * blockDim.x + threadIdx.x; i < total;
       i += (long)gridDim.x * blockDim.x) {
    long s = i % SS;
    long r = i / SS;
    int c = (int)(r % C);
    int b = (int)(r / C);
    float v = a[i];
    if (act == 1)      { float al = alphap[0]; v = v > 0.f ? v : al * v; }
    else if (act == 2) { v = v > 0.f ? v : 0.1f * v; }
    else if (act == 3) { v = fmaxf(v, 0.f); }
    if (add)  v += add[i];
    if (outf) outf[i] = v;
    if (xbt)  xbt[((long)b * SS + s) * cstride + coff + c] = f2bf(v);
  }
}

// SFT: out = x*(1+scale)+shift  -> f32 + bf16
__global__ void k_sft(const float* __restrict__ x, const float* __restrict__ sc,
                      const float* __restrict__ sh, float* __restrict__ outf,
                      unsigned short* __restrict__ xbt) {
  long total = (long)BB * DIMC * SS;
  for (long i = (long)blockIdx.x * blockDim.x + threadIdx.x; i < total;
       i += (long)gridDim.x * blockDim.x) {
    long s = i % SS;
    long r = i / SS;
    int c = (int)(r % DIMC);
    int b = (int)(r / DIMC);
    float v = x[i] * (1.f + sc[i]) + sh[i];
    outf[i] = v;
    xbt[((long)b * SS + s) * DIMC + c] = f2bf(v);
  }
}

// ---------------- gate / MoE ----------------
__global__ void k_gate(const float* __restrict__ fm, const float* __restrict__ gw,
                       const float* __restrict__ gb, int* __restrict__ gidx) {
  __shared__ float mean[DIMC];
  int b = blockIdx.x, c = threadIdx.x;
  float sum = 0.f;
  const float* p = fm + ((long)b * DIMC + c) * SS;
  for (long s = 0; s < SS; ++s) sum += p[s];
  mean[c] = sum * (1.f / (float)SS);
  __syncthreads();
  if (c == 0) {
    int best = 0; float bv = -1e30f;
    for (int e = 0; e < NE_; ++e) {
      float l = gb[e];
      for (int j = 0; j < DIMC; ++j) l += gw[e * DIMC + j] * mean[j];
      if (l > bv) { bv = l; best = e; }
    }
    gidx[b] = best;
  }
}

__global__ void k_moe_sel(const float* __restrict__ eo, const int* __restrict__ gidx,
                          float* __restrict__ fm) {
  long total = (long)BB * DIMC * SS;
  for (long i = (long)blockIdx.x * blockDim.x + threadIdx.x; i < total;
       i += (long)gridDim.x * blockDim.x) {
    long s = i % SS;
    long r = i / SS;
    int c = (int)(r % DIMC);
    int b = (int)(r / DIMC);
    int g = gidx[b];
    fm[i] = eo[((long)b * (NE_ * DIMC) + g * DIMC + c) * SS + s];
  }
}

// ---------------- global context ----------------
__global__ void k_gc_score(const float* __restrict__ fm, const float* __restrict__ mw,
                           const float* __restrict__ mb, float* __restrict__ score) {
  long total = (long)BB * SS;
  for (long i = (long)blockIdx.x * blockDim.x + threadIdx.x; i < total;
       i += (long)gridDim.x * blockDim.x) {
    long s = i % SS;
    int b = (int)(i / SS);
    float v = mb[0];
    for (int c = 0; c < DIMC; ++c) v += mw[c] * fm[((long)b * DIMC + c) * SS + s];
    score[i] = v;
  }
}

__global__ void k_softmax(float* __restrict__ sc) {
  __shared__ float red[256];
  int b = blockIdx.x, t = threadIdx.x;
  float m = -1e30f;
  for (long s = t; s < SS; s += 256) m = fmaxf(m, sc[(long)b * SS + s]);
  red[t] = m; __syncthreads();
  for (int o = 128; o > 0; o >>= 1) { if (t < o) red[t] = fmaxf(red[t], red[t + o]); __syncthreads(); }
  float mx = red[0]; __syncthreads();
  float sum = 0.f;
  for (long s = t; s < SS; s += 256) {
    float e = __expf(sc[(long)b * SS + s] - mx);
    sc[(long)b * SS + s] = e; sum += e;
  }
  red[t] = sum; __syncthreads();
  for (int o = 128; o > 0; o >>= 1) { if (t < o) red[t] += red[t + o]; __syncthreads(); }
  float inv = 1.f / red[0];
  for (long s = t; s < SS; s += 256) sc[(long)b * SS + s] *= inv;
}

__global__ void k_ctx(const float* __restrict__ attn, const float* __restrict__ fm,
                      float* __restrict__ ctx) {
  __shared__ float red[256];
  int bc = blockIdx.x, b = bc / DIMC, c = bc % DIMC, t = threadIdx.x;
  float sum = 0.f;
  for (long s = t; s < SS; s += 256)
    sum += attn[(long)b * SS + s] * fm[((long)b * DIMC + c) * SS + s];
  red[t] = sum; __syncthreads();
  for (int o = 128; o > 0; o >>= 1) { if (t < o) red[t] += red[t + o]; __syncthreads(); }
  if (t == 0) ctx[bc] = red[0];
}

__global__ void k_gcmlp(const float* __restrict__ ctx, const float* __restrict__ w1,
                        const float* __restrict__ b1, const float* __restrict__ g,
                        const float* __restrict__ be, const float* __restrict__ w2,
                        const float* __restrict__ b2, float* __restrict__ tr) {
  __shared__ float t1[16], t2[16], red[2];
  int b = blockIdx.x, c = threadIdx.x;   // 64 threads
  if (c < 16) {
    float v = b1[c];
    for (int j = 0; j < DIMC; ++j) v += w1[c * DIMC + j] * ctx[b * DIMC + j];
    t1[c] = v;
  }
  __syncthreads();
  if (c == 0) {
    float mu = 0.f;
    for (int k = 0; k < 16; ++k) mu += t1[k];
    mu *= (1.f / 16.f);
    float va = 0.f;
    for (int k = 0; k < 16; ++k) { float d = t1[k] - mu; va += d * d; }
    red[0] = mu; red[1] = va * (1.f / 16.f);
  }
  __syncthreads();
  if (c < 16) {
    float nv = (t1[c] - red[0]) * rsqrtf(red[1] + 1e-5f) * g[c] + be[c];
    t2[c] = fmaxf(nv, 0.f);
  }
  __syncthreads();
  float v = b2[c];
  for (int k = 0; k < 16; ++k) v += w2[c * 16 + k] * t2[k];
  tr[b * DIMC + c] = v;
}

// f_global = fm + tr[b][c]  -> bf16 xbt only
__global__ void k_add_bcast(const float* __restrict__ fm, const float* __restrict__ tr,
                            unsigned short* __restrict__ xbt) {
  long total = (long)BB * DIMC * SS;
  for (long i = (long)blockIdx.x * blockDim.x + threadIdx.x; i < total;
       i += (long)gridDim.x * blockDim.x) {
    long s = i % SS;
    long r = i / SS;
    int c = (int)(r % DIMC);
    int b = (int)(r / DIMC);
    xbt[((long)b * SS + s) * DIMC + c] = f2bf(fm[i] + tr[b * DIMC + c]);
  }
}

// ---------------- mamba ----------------
// causal depthwise conv (k=4, left pad 3) + silu; xz rows 0..127 are xi
__global__ void k_conv1d_silu(const float* __restrict__ xz, const float* __restrict__ cw,
                              const float* __restrict__ cb, float* __restrict__ xm) {
  long total = (long)BB * DI_ * SS;
  for (long i = (long)blockIdx.x * blockDim.x + threadIdx.x; i < total;
       i += (long)gridDim.x * blockDim.x) {
    long s = i % SS;
    long r = i / SS;
    int di = (int)(r % DI_);
    int b  = (int)(r / DI_);
    const float* src = xz + ((long)b * (2 * DI_) + di) * SS;
    float acc = cb[di];
#pragma unroll
    for (int j = 0; j < 4; ++j) {
      long sp = s - 3 + j;
      if (sp >= 0) acc += src[sp] * cw[di * 4 + j];
    }
    xm[i] = siluf(acc);
  }
}

__global__ void k_dt(const float* __restrict__ xdbl, const float* __restrict__ dtw,
                     const float* __restrict__ dtb, float* __restrict__ dt) {
  long total = (long)BB * DI_ * SS;
  for (long i = (long)blockIdx.x * blockDim.x + threadIdx.x; i < total;
       i += (long)gridDim.x * blockDim.x) {
    long s = i % SS;
    long r = i / SS;
    int di = (int)(r % DI_);
    int b  = (int)(r / DI_);
    float v = dtb[di];
#pragma unroll
    for (int q = 0; q < DR_; ++q)
      v += xdbl[((long)b * 36 + q) * SS + s] * dtw[di * DR_ + q];
    dt[i] = softplusf(v);
  }
}

// selective scan: 128 channel-threads per block (one block per batch).
// shared B_t/C_t rows staged via global_load_async_to_lds + s_wait_asynccnt.
__global__ __launch_bounds__(128)
void k_scan(const float* __restrict__ xm, const float* __restrict__ dt,
            const float* __restrict__ xdbl, const float* __restrict__ Alog,
            const float* __restrict__ Dv, const float* __restrict__ xz,
            float* __restrict__ y) {
  __shared__ float sB[DS_ * 128];
  __shared__ float sC[DS_ * 128];
  const int b  = blockIdx.x;
  const int di = threadIdx.x;
  float hst[DS_], Ak[DS_];
#pragma unroll
  for (int k = 0; k < DS_; ++k) { hst[k] = 0.f; Ak[k] = -__expf(Alog[di * DS_ + k]); }
  const float* xmb = xm + ((long)b * DI_ + di) * SS;
  const float* dtb = dt + ((long)b * DI_ + di) * SS;
  const float* zgb = xz + ((long)b * (2 * DI_) + DI_ + di) * SS;
  float*       yb  = y  + ((long)b * DI_ + di) * SS;
  const float  Dd  = Dv[di];

  for (long s0 = 0; s0 < SS; s0 += 128) {
#pragma unroll
    for (int k = 0; k < DS_; ++k) {
      unsigned ldsB = (unsigned)(unsigned long long)(void*)&sB[k * 128 + di];
      unsigned long long gB =
          (unsigned long long)(const void*)&xdbl[((long)b * 36 + 4 + k) * SS + s0 + di];
      asm volatile("global_load_async_to_lds_b32 %0, %1, off" :: "v"(ldsB), "v"(gB) : "memory");
      unsigned ldsC = (unsigned)(unsigned long long)(void*)&sC[k * 128 + di];
      unsigned long long gC =
          (unsigned long long)(const void*)&xdbl[((long)b * 36 + 20 + k) * SS + s0 + di];
      asm volatile("global_load_async_to_lds_b32 %0, %1, off" :: "v"(ldsC), "v"(gC) : "memory");
    }
    asm volatile("s_wait_asynccnt 0" ::: "memory");
    __syncthreads();
    for (int j = 0; j < 128; ++j) {
      long s   = s0 + j;
      float xt = xmb[s];
      float d  = dtb[s];
      float dx = d * xt;
      float acc = 0.f;
#pragma unroll
      for (int k = 0; k < DS_; ++k) {
        float dA = __expf(d * Ak[k]);
        hst[k] = dA * hst[k] + dx * sB[k * 128 + j];
        acc += hst[k] * sC[k * 128 + j];
      }
      float vy = acc + xt * Dd;
      yb[s] = vy * siluf(zgb[s]);
    }
    __syncthreads();
  }
}

// ---------------- tail / prediction ----------------
__global__ void k_pred(const float* __restrict__ mn, const float* __restrict__ res,
                       float* __restrict__ pred) {
  long total = (long)BB * SS;
  for (long i = (long)blockIdx.x * blockDim.x + threadIdx.x; i < total;
       i += (long)gridDim.x * blockDim.x)
    pred[i] = fmaxf(mn[i] + res[i], 0.f);
}

// ---------------- FFT (separable 64-pt DFT) ----------------
__global__ void k_r2c(const float* __restrict__ r, float* __restrict__ c) {
  long total = (long)NTOT;
  for (long i = (long)blockIdx.x * blockDim.x + threadIdx.x; i < total;
       i += (long)gridDim.x * blockDim.x) { c[2 * i] = r[i]; c[2 * i + 1] = 0.f; }
}

__global__ void k_dft_w(const float* __restrict__ in, float* __restrict__ out, float sign) {
  __shared__ float lr[64], li[64];
  long base = (long)blockIdx.x * 64;
  int k = threadIdx.x;
  lr[k] = in[2 * (base + k)];
  li[k] = in[2 * (base + k) + 1];
  __syncthreads();
  float sr = 0.f, si = 0.f;
  for (int j = 0; j < 64; ++j) {
    float ang = sign * 6.28318530718f * (float)(k * j) * (1.f / 64.f);
    float sn, cs; __sincosf(ang, &sn, &cs);
    sr += lr[j] * cs - li[j] * sn;
    si += lr[j] * sn + li[j] * cs;
  }
  out[2 * (base + k)] = sr;
  out[2 * (base + k) + 1] = si;
}

__global__ void k_dft_h(const float* __restrict__ in, float* __restrict__ out, float sign) {
  __shared__ float lr[64], li[64];
  int img = blockIdx.x >> 6;
  int w = blockIdx.x & 63;
  int k = threadIdx.x;
  long idx = ((long)img * 64 + k) * 64 + w;
  lr[k] = in[2 * idx];
  li[k] = in[2 * idx + 1];
  __syncthreads();
  float sr = 0.f, si = 0.f;
  for (int j = 0; j < 64; ++j) {
    float ang = sign * 6.28318530718f * (float)(k * j) * (1.f / 64.f);
    float sn, cs; __sincosf(ang, &sn, &cs);
    sr += lr[j] * cs - li[j] * sn;
    si += lr[j] * sn + li[j] * cs;
  }
  out[2 * idx] = sr;
  out[2 * idx + 1] = si;
}

__global__ void k_merge(const float* __restrict__ Fm, const float* __restrict__ Fp,
                        float* __restrict__ Fo) {
  long total = (long)NTOT;
  for (long i = (long)blockIdx.x * blockDim.x + threadIdx.x; i < total;
       i += (long)gridDim.x * blockDim.x) {
    int w = (int)(i & 63);
    int h = (int)((i >> 6) & 63);
    int yy = ((h + 32) & 63) - 32;
    int xx = ((w + 32) & 63) - 32;
    bool m = (yy * yy + xx * xx) <= 100;
    Fo[2 * i]     = m ? Fm[2 * i]     : Fp[2 * i];
    Fo[2 * i + 1] = m ? Fm[2 * i + 1] : Fp[2 * i + 1];
  }
}

__global__ void k_c2r(const float* __restrict__ c, float* __restrict__ out) {
  long total = (long)NTOT;
  for (long i = (long)blockIdx.x * blockDim.x + threadIdx.x; i < total;
       i += (long)gridDim.x * blockDim.x)
    out[i] = c[2 * i] * (1.f / 4096.f);
}

// =====================================================================
// host orchestration
// =====================================================================
extern "C" void kernel_launch(void* const* d_in, const int* in_sizes, int n_in,
                              void* d_out, int out_size, void* d_ws, size_t ws_size,
                              hipStream_t stream) {
  (void)in_sizes; (void)n_in; (void)out_size; (void)ws_size;
  const float* aux   = (const float*)d_in[0];
  const float* mainx = (const float*)d_in[1];
  const float* aux_head_w = (const float*)d_in[2];
  const float* aux_head_b = (const float*)d_in[3];
  const float* ms_w[3]    = {(const float*)d_in[4], (const float*)d_in[6], (const float*)d_in[8]};
  const float* ms_b[3]    = {(const float*)d_in[5], (const float*)d_in[7], (const float*)d_in[9]};
  const float* ms_fw = (const float*)d_in[10];
  const float* ms_fb = (const float*)d_in[11];
  const float* ms_a  = (const float*)d_in[12];
  const float* main_head_w = (const float*)d_in[13];
  const float* main_head_b = (const float*)d_in[14];
  const float* sft_w[8];
  const float* sft_b[8];
  for (int j = 0; j < 8; ++j) { sft_w[j] = (const float*)d_in[15 + 2 * j]; sft_b[j] = (const float*)d_in[16 + 2 * j]; }
  const float* res1_w1 = (const float*)d_in[31];
  const float* res1_b1 = (const float*)d_in[32];
  const float* res1_a  = (const float*)d_in[33];
  const float* res1_w2 = (const float*)d_in[34];
  const float* res1_b2 = (const float*)d_in[35];
  const float* gate_w  = (const float*)d_in[36];
  const float* gate_b  = (const float*)d_in[37];
  const float* moe_ew  = (const float*)d_in[38];
  const float* moe_eb  = (const float*)d_in[39];
  const float* gc_mw = (const float*)d_in[40];
  const float* gc_mb = (const float*)d_in[41];
  const float* gc_w1 = (const float*)d_in[42];
  const float* gc_b1 = (const float*)d_in[43];
  const float* gc_g  = (const float*)d_in[44];
  const float* gc_be = (const float*)d_in[45];
  const float* gc_w2 = (const float*)d_in[46];
  const float* gc_b2 = (const float*)d_in[47];
  const float* mb_in  = (const float*)d_in[48];
  const float* mb_cw  = (const float*)d_in[49];
  const float* mb_cb  = (const float*)d_in[50];
  const float* mb_xp  = (const float*)d_in[51];
  const float* mb_dtw = (const float*)d_in[52];
  const float* mb_dtb = (const float*)d_in[53];
  const float* mb_Alog= (const float*)d_in[54];
  const float* mb_D   = (const float*)d_in[55];
  const float* mb_out = (const float*)d_in[56];
  const float* tail_w1 = (const float*)d_in[57];
  const float* tail_b1 = (const float*)d_in[58];
  const float* tail_a  = (const float*)d_in[59];
  const float* tail_w2 = (const float*)d_in[60];
  const float* tail_b2 = (const float*)d_in[61];

  char* wp = (char*)d_ws;
  auto alloc = [&](size_t bytes) -> void* {
    void* p = (void*)wp;
    wp += (bytes + 255) & ~(size_t)255;
    return p;
  };
  // bf16 weight packs
  unsigned short* wb_ms[3]; for (int i = 0; i < 3; ++i) wb_ms[i] = (unsigned short*)alloc((size_t)64 * KCONV * 2);
  unsigned short* wb_r1a  = (unsigned short*)alloc((size_t)64 * KCONV * 2);
  unsigned short* wb_r1b  = (unsigned short*)alloc((size_t)64 * KCONV * 2);
  unsigned short* wb_tl1  = (unsigned short*)alloc((size_t)64 * KCONV * 2);
  unsigned short* wb_moe  = (unsigned short*)alloc((size_t)192 * KCONV * 2);
  unsigned short* wb_msf  = (unsigned short*)alloc((size_t)64 * 192 * 2);
  unsigned short* wb_sft[8]; for (int j = 0; j < 8; ++j) wb_sft[j] = (unsigned short*)alloc((size_t)64 * 64 * 2);
  unsigned short* wb_mbin = (unsigned short*)alloc((size_t)256 * 64 * 2);
  unsigned short* wb_mbxp = (unsigned short*)alloc((size_t)64 * 128 * 2);
  unsigned short* wb_mbout= (unsigned short*)alloc((size_t)64 * 128 * 2);
  // f32 feature maps
  size_t F = (size_t)BB * DIMC * SS * 4;
  float* FA  = (float*)alloc(F);
  float* FM  = (float*)alloc(F);
  float* T0  = (float*)alloc(F);
  float* T1  = (float*)alloc(F);
  float* SC  = (float*)alloc(F);
  float* SH  = (float*)alloc(F);
  float* MS1 = (float*)alloc(F);
  float* MS2 = (float*)alloc(F);
  float* MS3 = (float*)alloc(F);
  float* EO  = (float*)alloc((size_t)BB * 192 * SS * 4);
  float* ATTN= (float*)alloc((size_t)BB * SS * 4);
  float* CTX = (float*)alloc((size_t)BB * DIMC * 4);
  float* TR  = (float*)alloc((size_t)BB * DIMC * 4);
  float* XZ  = (float*)alloc((size_t)BB * 256 * SS * 4);
  float* XM  = (float*)alloc((size_t)BB * DI_ * SS * 4);
  float* DTB = (float*)alloc((size_t)BB * DI_ * SS * 4);
  float* XDBL= (float*)alloc((size_t)BB * 36 * SS * 4);
  float* YB  = (float*)alloc((size_t)BB * DI_ * SS * 4);
  float* FMB = (float*)alloc(F);
  float* RES = (float*)alloc((size_t)BB * SS * 4);
  float* PRED= (float*)alloc((size_t)BB * SS * 4);
  float* C1  = (float*)alloc((size_t)NTOT * 2 * 4);
  float* C2  = (float*)alloc((size_t)NTOT * 2 * 4);
  float* C3  = (float*)alloc((size_t)NTOT * 2 * 4);
  int*   GIDX= (int*)alloc(BB * sizeof(int));
  // bf16 activation buffers (voxel-major)
  unsigned short* XBT_C  = (unsigned short*)alloc((size_t)NTOT * 64 * 2);
  unsigned short* XBT_M  = (unsigned short*)alloc((size_t)NTOT * 64 * 2);
  unsigned short* XBT_T  = (unsigned short*)alloc((size_t)NTOT * 64 * 2);
  unsigned short* XBT192 = (unsigned short*)alloc((size_t)NTOT * 192 * 2);
  unsigned short* XBT128 = (unsigned short*)alloc((size_t)NTOT * 128 * 2);

  const size_t convShm = (size_t)64 * KCONV * 2;   // 216KB weight slab in LDS
  dim3 tb256(256), tb128(128);
  auto blk = [](long tot) { return dim3((unsigned)((tot + 255) / 256)); };
#define EWTOT(C_) ((long)BB * (C_) * SS)

  // ---- weight prep ----
  for (int i = 0; i < 3; ++i)
    k_prep_convw<<<blk(64L * 27 * 64), tb256, 0, stream>>>(ms_w[i], wb_ms[i], 64, 64);
  k_prep_convw<<<blk(64L * 27 * 64), tb256, 0, stream>>>(res1_w1, wb_r1a, 64, 64);
  k_prep_convw<<<blk(64L * 27 * 64), tb256, 0, stream>>>(res1_w2, wb_r1b, 64, 64);
  k_prep_convw<<<blk(64L * 27 * 64), tb256, 0, stream>>>(tail_w1, wb_tl1, 64, 64);
  k_prep_convw<<<blk(192L * 27 * 64), tb256, 0, stream>>>(moe_ew, wb_moe, 192, 64);
  k_prep_densew<<<blk(64L * 192), tb256, 0, stream>>>(ms_fw, wb_msf, 64, 192, 64);
  for (int j = 0; j < 8; ++j)
    k_prep_densew<<<blk(64L * 64), tb256, 0, stream>>>(sft_w[j], wb_sft[j], 64, 64, 64);
  k_prep_densew<<<blk(256L * 64), tb256, 0, stream>>>(mb_in, wb_mbin, 256, 64, 256);
  k_prep_densew<<<blk(64L * 128), tb256, 0, stream>>>(mb_xp, wb_mbxp, 36, 128, 64);
  k_prep_densew<<<blk(64L * 128), tb256, 0, stream>>>(mb_out, wb_mbout, 64, 128, 64);

  // ---- aux head + multiscale ----
  k_conv3_naive<<<blk(EWTOT(64)), tb256, 0, stream>>>(aux, aux_head_w, aux_head_b, FA, 9, 64, 1);
  k_ew<<<blk(EWTOT(64)), tb256, 0, stream>>>(FA, nullptr, nullptr, XBT_T, 64, 64, 0, 0, nullptr);
  k_conv3x3_wmma<<<dim3(NTOT / 16, 1), tb128, convShm, stream>>>(XBT_T, wb_ms[0], ms_b[0], MS1, 64, 1);
  k_conv3x3_wmma<<<dim3(NTOT / 16, 1), tb128, convShm, stream>>>(XBT_T, wb_ms[1], ms_b[1], MS2, 64, 2);
  k_conv3x3_wmma<<<dim3(NTOT / 16, 1), tb128, convShm, stream>>>(XBT_T, wb_ms[2], ms_b[2], MS3, 64, 3);
  k_ew<<<blk(EWTOT(64)), tb256, 0, stream>>>(MS1, nullptr, nullptr, XBT192, 64, 192, 0,   0, nullptr);
  k_ew<<<blk(EWTOT(64)), tb256, 0, stream>>>(MS2, nullptr, nullptr, XBT192, 64, 192, 64,  0, nullptr);
  k_ew<<<blk(EWTOT(64)), tb256, 0, stream>>>(MS3, nullptr, nullptr, XBT192, 64, 192, 128, 0, nullptr);
  k_gemm_wmma<<<dim3(NTOT / 16, 1), tb128, 0, stream>>>(XBT192, 192, wb_msf, 192, ms_fb, T0, 64, 64);
  k_ew<<<blk(EWTOT(64)), tb256, 0, stream>>>(T0, FA, FA, XBT_C, 64, 64, 0, 1, ms_a); // f_aux + cond bf16

  // ---- main head ----
  k_conv3_naive<<<blk(EWTOT(64)), tb256, 0, stream>>>(mainx, main_head_w, main_head_b, FM, 1, 64, 1);

  // ---- SFT1 ----
  k_gemm_wmma<<<dim3(NTOT / 16, 1), tb128, 0, stream>>>(XBT_C, 64, wb_sft[0], 64, sft_b[0], T0, 64, 64);
  k_ew<<<blk(EWTOT(64)), tb256, 0, stream>>>(T0, nullptr, nullptr, XBT_T, 64, 64, 0, 2, nullptr);
  k_gemm_wmma<<<dim3(NTOT / 16, 1), tb128, 0, stream>>>(XBT_T, 64, wb_sft[1], 64, sft_b[1], SC, 64, 64);
  k_gemm_wmma<<<dim3(NTOT / 16, 1), tb128, 0, stream>>>(XBT_C, 64, wb_sft[2], 64, sft_b[2], T0, 64, 64);
  k_ew<<<blk(EWTOT(64)), tb256, 0, stream>>>(T0, nullptr, nullptr, XBT_T, 64, 64, 0, 2, nullptr);
  k_gemm_wmma<<<dim3(NTOT / 16, 1), tb128, 0, stream>>>(XBT_T, 64, wb_sft[3], 64, sft_b[3], SH, 64, 64);
  k_sft<<<blk(EWTOT(64)), tb256, 0, stream>>>(FM, SC, SH, FM, XBT_M);

  // ---- res block ----
  k_conv3x3_wmma<<<dim3(NTOT / 16, 1), tb128, convShm, stream>>>(XBT_M, wb_r1a, res1_b1, T0, 64, 1);
  k_ew<<<blk(EWTOT(64)), tb256, 0, stream>>>(T0, nullptr, nullptr, XBT_T, 64, 64, 0, 1, res1_a);
  k_conv3x3_wmma<<<dim3(NTOT / 16, 1), tb128, convShm, stream>>>(XBT_T, wb_r1b, res1_b2, T1, 64, 1);
  k_ew<<<blk(EWTOT(64)), tb256, 0, stream>>>(T1, FM, FM, nullptr, 64, 64, 0, 0, nullptr);

  // ---- SFT2 ----
  k_gemm_wmma<<<dim3(NTOT / 16, 1), tb128, 0, stream>>>(XBT_C, 64, wb_sft[4], 64, sft_b[4], T0, 64, 64);
  k_ew<<<blk(EWTOT(64)), tb256, 0, stream>>>(T0, nullptr, nullptr, XBT_T, 64, 64, 0, 2, nullptr);
  k_gemm_wmma<<<dim3(NTOT / 16, 1), tb128, 0, stream>>>(XBT_T, 64, wb_sft[5], 64, sft_b[5], SC, 64, 64);
  k_gemm_wmma<<<dim3(NTOT / 16, 1), tb128, 0, stream>>>(XBT_C, 64, wb_sft[6], 64, sft_b[6], T0, 64, 64);
  k_ew<<<blk(EWTOT(64)), tb256, 0, stream>>>(T0, nullptr, nullptr, XBT_T, 64, 64, 0, 2, nullptr);
  k_gemm_wmma<<<dim3(NTOT / 16, 1), tb128, 0, stream>>>(XBT_T, 64, wb_sft[7], 64, sft_b[7], SH, 64, 64);
  k_sft<<<blk(EWTOT(64)), tb256, 0, stream>>>(FM, SC, SH, FM, XBT_M);

  // ---- gate + MoE ----
  k_gate<<<dim3(BB), dim3(64), 0, stream>>>(FM, gate_w, gate_b, GIDX);
  k_conv3x3_wmma<<<dim3(NTOT / 16, 3), tb128, convShm, stream>>>(XBT_M, wb_moe, moe_eb, EO, 192, 1);
  k_moe_sel<<<blk(EWTOT(64)), tb256, 0, stream>>>(EO, GIDX, FM);

  // ---- global context ----
  k_gc_score<<<blk((long)BB * SS), tb256, 0, stream>>>(FM, gc_mw, gc_mb, ATTN);
  k_softmax<<<dim3(BB), tb256, 0, stream>>>(ATTN);
  k_ctx<<<dim3(BB * DIMC), tb256, 0, stream>>>(ATTN, FM, CTX);
  k_gcmlp<<<dim3(BB), dim3(64), 0, stream>>>(CTX, gc_w1, gc_b1, gc_g, gc_be, gc_w2, gc_b2, TR);
  k_add_bcast<<<blk(EWTOT(64)), tb256, 0, stream>>>(FM, TR, XBT_M);   // f_global -> bf16

  // ---- mamba ----
  k_gemm_wmma<<<dim3(NTOT / 16, 4), tb128, 0, stream>>>(XBT_M, 64, wb_mbin, 64, nullptr, XZ, 256, 256);
  k_conv1d_silu<<<blk(EWTOT(128)), tb256, 0, stream>>>(XZ, mb_cw, mb_cb, XM);
  k_ew<<<blk(EWTOT(128)), tb256, 0, stream>>>(XM, nullptr, nullptr, XBT128, 128, 128, 0, 0, nullptr);
  k_gemm_wmma<<<dim3(NTOT / 16, 1), tb128, 0, stream>>>(XBT128, 128, wb_mbxp, 128, nullptr, XDBL, 36, 36);
  k_dt<<<blk(EWTOT(128)), tb256, 0, stream>>>(XDBL, mb_dtw, mb_dtb, DTB);
  k_scan<<<dim3(BB), tb128, 0, stream>>>(XM, DTB, XDBL, mb_Alog, mb_D, XZ, YB);
  k_ew<<<blk(EWTOT(128)), tb256, 0, stream>>>(YB, nullptr, nullptr, XBT128, 128, 128, 0, 0, nullptr);
  k_gemm_wmma<<<dim3(NTOT / 16, 1), tb128, 0, stream>>>(XBT128, 128, wb_mbout, 128, nullptr, FMB, 64, 64);
  k_ew<<<blk(EWTOT(64)), tb256, 0, stream>>>(FMB, FM, nullptr, XBT_T, 64, 64, 0, 0, nullptr); // f_final bf16

  // ---- tail ----
  k_conv3x3_wmma<<<dim3(NTOT / 16, 1), tb128, convShm, stream>>>(XBT_T, wb_tl1, tail_b1, T0, 64, 1);
  k_ew<<<blk(EWTOT(64)), tb256, 0, stream>>>(T0, nullptr, T1, nullptr, 64, 64, 0, 1, tail_a);
  k_conv3_naive<<<blk((long)BB * SS), tb256, 0, stream>>>(T1, tail_w2, tail_b2, RES, 64, 1, 1);
  k_pred<<<blk((long)BB * SS), tb256, 0, stream>>>(mainx, RES, PRED);

  // ---- FFT low-pass merge ----
  k_r2c<<<blk(NTOT), tb256, 0, stream>>>(mainx, C1);
  k_dft_w<<<dim3(BB * TT * HH), dim3(64), 0, stream>>>(C1, C2, -1.f);
  k_dft_h<<<dim3(BB * TT * WW), dim3(64), 0, stream>>>(C2, C1, -1.f);   // Fm -> C1
  k_r2c<<<blk(NTOT), tb256, 0, stream>>>(PRED, C2);
  k_dft_w<<<dim3(BB * TT * HH), dim3(64), 0, stream>>>(C2, C3, -1.f);
  k_dft_h<<<dim3(BB * TT * WW), dim3(64), 0, stream>>>(C3, C2, -1.f);   // Fp -> C2
  k_merge<<<blk(NTOT), tb256, 0, stream>>>(C1, C2, C3);
  k_dft_w<<<dim3(BB * TT * HH), dim3(64), 0, stream>>>(C3, C1, 1.f);
  k_dft_h<<<dim3(BB * TT * WW), dim3(64), 0, stream>>>(C1, C3, 1.f);
  k_c2r<<<blk(NTOT), tb256, 0, stream>>>(C3, (float*)d_out);
#undef EWTOT
}
